// Sinkhorn_40973988004403
// MI455X (gfx1250) — compile-verified, matrix operand order checked by reading, not run on hardware
//
#include <hip/hip_runtime.h>

#define NB     4096      // B (rows of both p_s and p_t)
#define NV     4096      // V (feature dim) == K of GEMM
#define INV_T        0.5f
#define INV_EPSILON  10.0f
#define EPS_C        1e-5f
#define LOSS_SCALE_C 0.0008f
#define SINK_ITERS   20

typedef __attribute__((ext_vector_type(16))) _Float16 v16h;
typedef __attribute__((ext_vector_type(8)))  float    v8f;
typedef __attribute__((ext_vector_type(4)))  float    f32x4;

union HFrag { v16h v; f32x4 q[2]; };

// 16-bit A/B fragment, 16x32 (MxK): lane L holds row (L&15); lanes 0-15 carry
// K = k0..k0+7 and k0+16..k0+23, lanes 16-31 carry K = k0+8.. and k0+24..
// -> two contiguous 16-byte loads per lane at (k + (L>>4)*8) and +16.
static __device__ __forceinline__ v16h load_frag16(const _Float16* rowBase, int k0) {
    HFrag f;
    f.q[0] = *(const f32x4*)(rowBase + k0);
    f.q[1] = *(const f32x4*)(rowBase + k0 + 16);
    return f.v;
}

// ---------------------------------------------------------------- softmax ---
// One block per row: p = softmax(y/T) stored as f16 for WMMA, plus
// sum(p^2) (f32) per row for the Poincare norm terms.
__global__ void __launch_bounds__(256)
softmax_rows_k(const float* __restrict__ y, _Float16* __restrict__ p,
               float* __restrict__ nrm) {
    __shared__ float red[256];
    const int tid = threadIdx.x;
    const int row = blockIdx.x;
    const float* yr = y + (size_t)row * NV;

    float v[16];
    float m = -3.402823466e38f;
    #pragma unroll
    for (int e = 0; e < 16; ++e) { v[e] = yr[e * 256 + tid]; m = fmaxf(m, v[e]); }
    red[tid] = m; __syncthreads();
    for (int o = 128; o > 0; o >>= 1) {
        if (tid < o) red[tid] = fmaxf(red[tid], red[tid + o]);
        __syncthreads();
    }
    m = red[0]; __syncthreads();

    float s = 0.f;
    #pragma unroll
    for (int e = 0; e < 16; ++e) { float t = expf((v[e] - m) * INV_T); v[e] = t; s += t; }
    red[tid] = s; __syncthreads();
    for (int o = 128; o > 0; o >>= 1) {
        if (tid < o) red[tid] += red[tid + o];
        __syncthreads();
    }
    const float inv = 1.f / red[0]; __syncthreads();

    float q = 0.f;
    _Float16* pr = p + (size_t)row * NV;
    #pragma unroll
    for (int e = 0; e < 16; ++e) {
        float pe = v[e] * inv;
        pr[e * 256 + tid] = (_Float16)pe;
        q += pe * pe;
    }
    red[tid] = q; __syncthreads();
    for (int o = 128; o > 0; o >>= 1) {
        if (tid < o) red[tid] += red[tid + o];
        __syncthreads();
    }
    if (tid == 0) nrm[row] = red[0];
}

__global__ void init_ones_k(float* __restrict__ v) {
    v[blockIdx.x * 256 + threadIdx.x] = 1.f;
}

// ------------------------------------------------- WMMA GEMM + Poincare ----
// C = p_s . p_t^T via V_WMMA_F32_16X16X32_F16; epilogue fuses the full
// Poincare distance + K = exp(-W/eps). 128x128 tile per block; 8 waves in a
// 2x4 grid, each wave owns a 64x32 patch = 4x2 WMMA accumulators.
// Ping-pong pipelined k-loop: two fixed fragment sets alternate between
// "being consumed by WMMA" and "being reloaded", so no register rotation
// copies are needed and ~12 global_load_b128 stay in flight under the WMMAs.
__global__ void __launch_bounds__(256)
poincare_gemm_k(const _Float16* __restrict__ pS, const _Float16* __restrict__ pT,
                const float* __restrict__ xn, const float* __restrict__ yn,
                float* __restrict__ W, float* __restrict__ Km) {
    const int tid   = threadIdx.x;
    const int lane  = tid & 31;
    const int wave  = tid >> 5;
    const int wy    = wave >> 2;              // 0..1
    const int wx    = wave & 3;               // 0..3
    const int rowBase = blockIdx.y * 128 + wy * 64;
    const int colBase = blockIdx.x * 128 + wx * 32;
    const int lr    = lane & 15;
    const int khalf = (lane >> 4) << 3;

    const _Float16* aRow[4];
    #pragma unroll
    for (int mt = 0; mt < 4; ++mt)
        aRow[mt] = pS + (size_t)(rowBase + mt * 16 + lr) * NV;
    const _Float16* bRow[2];
    #pragma unroll
    for (int nt = 0; nt < 2; ++nt)
        bRow[nt] = pT + (size_t)(colBase + nt * 16 + lr) * NV;

    v8f zero = {};
    v8f acc[4][2];
    #pragma unroll
    for (int mt = 0; mt < 4; ++mt)
        #pragma unroll
        for (int nt = 0; nt < 2; ++nt) acc[mt][nt] = zero;

    // Prologue: set A holds k=0, set B holds k=32.
    v16h afA[4], bfA[2], afB[4], bfB[2];
    #pragma unroll
    for (int mt = 0; mt < 4; ++mt) afA[mt] = load_frag16(aRow[mt], khalf);
    #pragma unroll
    for (int nt = 0; nt < 2; ++nt) bfA[nt] = load_frag16(bRow[nt], khalf);
    #pragma unroll
    for (int mt = 0; mt < 4; ++mt) afB[mt] = load_frag16(aRow[mt], 32 + khalf);
    #pragma unroll
    for (int nt = 0; nt < 2; ++nt) bfB[nt] = load_frag16(bRow[nt], 32 + khalf);

    #pragma unroll 1
    for (int k = 0; k < NV - 64; k += 64) {
        // Consume set A (k), then refill it for k+64 while set B is consumed.
        #pragma unroll
        for (int mt = 0; mt < 4; ++mt)
            #pragma unroll
            for (int nt = 0; nt < 2; ++nt)
                acc[mt][nt] = __builtin_amdgcn_wmma_f32_16x16x32_f16(
                    false, afA[mt], false, bfA[nt], (short)0, acc[mt][nt],
                    false, false);
        const int ka = k + 64 + khalf;
        #pragma unroll
        for (int mt = 0; mt < 4; ++mt) afA[mt] = load_frag16(aRow[mt], ka);
        #pragma unroll
        for (int nt = 0; nt < 2; ++nt) bfA[nt] = load_frag16(bRow[nt], ka);

        // Consume set B (k+32), then refill it for k+96.
        #pragma unroll
        for (int mt = 0; mt < 4; ++mt)
            #pragma unroll
            for (int nt = 0; nt < 2; ++nt)
                acc[mt][nt] = __builtin_amdgcn_wmma_f32_16x16x32_f16(
                    false, afB[mt], false, bfB[nt], (short)0, acc[mt][nt],
                    false, false);
        const int kb = k + 96 + khalf;
        #pragma unroll
        for (int mt = 0; mt < 4; ++mt) afB[mt] = load_frag16(aRow[mt], kb);
        #pragma unroll
        for (int nt = 0; nt < 2; ++nt) bfB[nt] = load_frag16(bRow[nt], kb);
    }
    // Tail: set A holds k = NV-64, set B holds k = NV-32.
    #pragma unroll
    for (int mt = 0; mt < 4; ++mt)
        #pragma unroll
        for (int nt = 0; nt < 2; ++nt)
            acc[mt][nt] = __builtin_amdgcn_wmma_f32_16x16x32_f16(
                false, afA[mt], false, bfA[nt], (short)0, acc[mt][nt],
                false, false);
    #pragma unroll
    for (int mt = 0; mt < 4; ++mt)
        #pragma unroll
        for (int nt = 0; nt < 2; ++nt)
            acc[mt][nt] = __builtin_amdgcn_wmma_f32_16x16x32_f16(
                false, afB[mt], false, bfB[nt], (short)0, acc[mt][nt],
                false, false);

    // Fused Poincare epilogue. C/D layout: VGPR r -> M = r (lanes 0-15) or
    // M = 8+r (lanes 16-31); N = lane&15.
    #pragma unroll
    for (int mt = 0; mt < 4; ++mt) {
        const int iBase = rowBase + mt * 16 + ((lane >> 4) << 3);
        float xnv[8], xcv[8];
        #pragma unroll
        for (int r = 0; r < 8; ++r) {
            xnv[r] = xn[iBase + r];
            xcv[r] = fminf(fmaxf(xnv[r], 0.f), 1.f - EPS_C);
        }
        #pragma unroll
        for (int nt = 0; nt < 2; ++nt) {
            const int j   = colBase + nt * 16 + lr;
            const float ynj = yn[j];
            const float yc  = fminf(fmaxf(ynj, 0.f), 1.f - EPS_C);
            #pragma unroll
            for (int r = 0; r < 8; ++r) {
                const int i = iBase + r;
                const float dot = acc[mt][nt][r];
                const float sq  = fmaxf(xnv[r] + ynj - 2.f * dot, 0.f);
                const float den = fmaxf((1.f - xcv[r]) * (1.f - yc), EPS_C);
                const float alpha = fmaxf(2.f * sq / den, 1.f + EPS_C);
                const float d2 = acoshf(alpha);
                const float w  = sqrtf(fmaxf(d2, 0.f));
                const size_t idx = (size_t)i * NB + j;
                W[idx]  = w;
                Km[idx] = expf(-w * INV_EPSILON);
            }
        }
    }
}

// --------------------------------------------------------- Sinkhorn steps --
// a = 1 / (K b): one wave per row, b cached in LDS (b128 reads), 128B global
// vector loads of K.
__global__ void __launch_bounds__(256)
row_matvec_inv_k(const float* __restrict__ Km, const float* __restrict__ bv,
                 float* __restrict__ av) {
    __shared__ float bc[NV];
    const int tid = threadIdx.x;
    for (int t = tid; t < NV; t += 256) bc[t] = bv[t];
    __syncthreads();
    const int lane = tid & 31;
    const int row  = blockIdx.x * 8 + (tid >> 5);
    const float* Kr = Km + (size_t)row * NV;
    float acc = 0.f;
    for (int it = 0; it < NV / 128; ++it) {
        const int j = (it * 32 + lane) * 4;
        f32x4 kq = *(const f32x4*)(Kr + j);
        f32x4 bq = *(const f32x4*)(bc + j);      // ds_load_b128
        acc += kq[0] * bq[0] + kq[1] * bq[1] + kq[2] * bq[2] + kq[3] * bq[3];
    }
    #pragma unroll
    for (int o = 16; o > 0; o >>= 1) acc += __shfl_xor(acc, o, 32);
    if (lane == 0) av[row] = 1.f / acc;
}

// Partial K^T a: block (bx,by) reduces 128 rows into colpart[by][j]
// (deterministic two-stage column reduction — no float atomics).
__global__ void __launch_bounds__(256)
col_matvec_part_k(const float* __restrict__ Km, const float* __restrict__ av,
                  float* __restrict__ colpart) {
    const int j  = blockIdx.x * 256 + threadIdx.x;
    const int i0 = blockIdx.y * 128;
    const float* p = Km + (size_t)i0 * NV + j;
    float acc = 0.f;
    #pragma unroll 4
    for (int ii = 0; ii < 128; ++ii)
        acc += p[(size_t)ii * NV] * av[i0 + ii];
    colpart[blockIdx.y * NV + j] = acc;
}

// b = 1 / (K^T a): fold the 32 partials in fixed order.
__global__ void __launch_bounds__(256)
col_finalize_k(const float* __restrict__ colpart, float* __restrict__ bv) {
    const int j = blockIdx.x * 256 + threadIdx.x;
    float s = 0.f;
    #pragma unroll
    for (int c = 0; c < 32; ++c) s += colpart[c * NV + j];
    bv[j] = 1.f / s;
}

// ----------------------------------------------------------- final loss ----
// part[b] = sum over 8 rows of a_i * sum_j K_ij * b_j * W_ij
__global__ void __launch_bounds__(256)
loss_part_k(const float* __restrict__ Km, const float* __restrict__ W,
            const float* __restrict__ av, const float* __restrict__ bv,
            float* __restrict__ part) {
    __shared__ float bc[NV];
    __shared__ float wred[8];
    const int tid = threadIdx.x;
    for (int t = tid; t < NV; t += 256) bc[t] = bv[t];
    __syncthreads();
    const int lane = tid & 31;
    const int wave = tid >> 5;
    const int row  = blockIdx.x * 8 + wave;
    const float* Kr = Km + (size_t)row * NV;
    const float* Wr = W  + (size_t)row * NV;
    float acc = 0.f;
    for (int it = 0; it < NV / 128; ++it) {
        const int j = (it * 32 + lane) * 4;
        f32x4 kq = *(const f32x4*)(Kr + j);
        f32x4 wq = *(const f32x4*)(Wr + j);
        f32x4 bq = *(const f32x4*)(bc + j);      // ds_load_b128
        acc += kq[0] * wq[0] * bq[0] + kq[1] * wq[1] * bq[1]
             + kq[2] * wq[2] * bq[2] + kq[3] * wq[3] * bq[3];
    }
    #pragma unroll
    for (int o = 16; o > 0; o >>= 1) acc += __shfl_xor(acc, o, 32);
    if (lane == 0) wred[wave] = av[row] * acc;
    __syncthreads();
    if (tid == 0) {
        float s = 0.f;
        #pragma unroll
        for (int w2 = 0; w2 < 8; ++w2) s += wred[w2];
        part[blockIdx.x] = s;
    }
}

__global__ void __launch_bounds__(256)
final_reduce_k(const float* __restrict__ part, float* __restrict__ out) {
    __shared__ float red[256];
    const int tid = threadIdx.x;
    red[tid] = part[tid] + part[tid + 256];
    __syncthreads();
    for (int o = 128; o > 0; o >>= 1) {
        if (tid < o) red[tid] += red[tid + o];
        __syncthreads();
    }
    if (tid == 0) out[0] = LOSS_SCALE_C * red[0];
}

// ---------------------------------------------------------------- driver ---
extern "C" void kernel_launch(void* const* d_in, const int* in_sizes, int n_in,
                              void* d_out, int out_size, void* d_ws, size_t ws_size,
                              hipStream_t stream) {
    (void)in_sizes; (void)n_in; (void)out_size; (void)ws_size;
    const float* y_s = (const float*)d_in[0];
    const float* y_t = (const float*)d_in[1];
    float* out = (float*)d_out;

    char* ws = (char*)d_ws;
    const size_t MB = 1ull << 20;
    _Float16* pS = (_Float16*)(ws);                 // 32 MB
    _Float16* pT = (_Float16*)(ws + 32 * MB);       // 32 MB
    float* W       = (float*)(ws + 64 * MB);        // 64 MB
    float* Km      = (float*)(ws + 128 * MB);       // 64 MB
    float* xn      = (float*)(ws + 192 * MB);       // 16 KB
    float* yn      = xn + NB;
    float* av      = yn + NB;
    float* bv      = av + NB;
    float* colpart = bv + NB;                       // 32 * 4096 floats
    float* part    = colpart + 32 * NV;             // 512 floats

    softmax_rows_k<<<NB, 256, 0, stream>>>(y_s, pS, xn);
    softmax_rows_k<<<NB, 256, 0, stream>>>(y_t, pT, yn);
    init_ones_k<<<NB / 256, 256, 0, stream>>>(bv);

    poincare_gemm_k<<<dim3(32, 32), 256, 0, stream>>>(pS, pT, xn, yn, W, Km);

    for (int it = 0; it < SINK_ITERS; ++it) {
        row_matvec_inv_k<<<NB / 8, 256, 0, stream>>>(Km, bv, av);
        col_matvec_part_k<<<dim3(16, 32), 256, 0, stream>>>(Km, av, colpart);
        col_finalize_k<<<16, 256, 0, stream>>>(colpart, bv);
    }

    loss_part_k<<<NB / 8, 256, 0, stream>>>(Km, W, av, bv, part);
    final_reduce_k<<<1, 256, 0, stream>>>(part, out);
}
